// CausalSelfAttention_60644938219761
// MI455X (gfx1250) — matrix operand order repaired and checked by reading, not compiled, not run on hardware
//
#include <hip/hip_runtime.h>
#include <hip/hip_bf16.h>

// ---------------- types ----------------
typedef __attribute__((ext_vector_type(16))) __bf16 v16bf;
typedef __attribute__((ext_vector_type(8)))  __bf16 v8bf;
typedef __attribute__((ext_vector_type(8)))  float  v8f;

#define B_SZ   2
#define S_LEN  2048
#define DMODEL 1024
#define NH     16
#define HD     64

__device__ __forceinline__ __bf16 f2bf(float x) {
  union { float f; unsigned u; } cv; cv.f = x;
  unsigned r = cv.u + 0x7FFFu + ((cv.u >> 16) & 1u);   // round-to-nearest-even
  unsigned short hs = (unsigned short)(r >> 16);
  __bf16 b; __builtin_memcpy(&b, &hs, 2);
  return b;
}

// concat two v8bf into the 16-element A/B fragment (pure register placement)
__device__ __forceinline__ v16bf mk_frag(v8bf lo, v8bf hi) {
  return __builtin_shufflevector(lo, hi, 0, 1, 2, 3, 4, 5, 6, 7,
                                         8, 9, 10, 11, 12, 13, 14, 15);
}

__device__ __forceinline__ v8f wmma_bf16(v16bf a, v16bf b, v8f c) {
  return __builtin_amdgcn_wmma_f32_16x16x32_bf16(false, a, false, b, (short)0, c,
                                                 false, false);
}

// CDNA5 async global->LDS copy, 16B per lane; tracked by ASYNCcnt.
// LDS aperture truncates flat addr to low 32 bits (ISA 10.2), so the low word of
// the generic pointer is the LDS byte address the instruction needs.
__device__ __forceinline__ void async_ld_b128(void* lds_dst, const void* gsrc) {
  unsigned lds = (unsigned)(size_t)lds_dst;
  asm volatile("global_load_async_to_lds_b128 %0, %1, off"
               :: "v"(lds), "v"(gsrc) : "memory");
}
__device__ __forceinline__ void wait_async0() {
  asm volatile("s_wait_asynccnt 0" ::: "memory");
}

// ---------------- prep kernels ----------------
__global__ __launch_bounds__(256) void cvt_f32_bf16(const float* __restrict__ in,
                                                    __bf16* __restrict__ out, unsigned n) {
  unsigned i = blockIdx.x * blockDim.x + threadIdx.x;
  if (i < n) out[i] = f2bf(in[i]);
}

// w[K][N] (f32) -> wt[N][K] (bf16)
__global__ __launch_bounds__(256) void transpose_f32_to_bf16(const float* __restrict__ w,
                                                             __bf16* __restrict__ wt,
                                                             int K, int N) {
  unsigned i = blockIdx.x * blockDim.x + threadIdx.x;
  unsigned total = (unsigned)K * (unsigned)N;
  if (i >= total) return;
  int k = i / N, n = i % N;
  wt[(size_t)n * K + k] = f2bf(w[i]);
}

// vt[b][h][d][s] = qkvh[b*S+s][2*D + h*64 + d]
__global__ __launch_bounds__(256) void build_vt(const __bf16* __restrict__ qkvh,
                                                __bf16* __restrict__ vt) {
  unsigned i = blockIdx.x * blockDim.x + threadIdx.x;
  unsigned total = (unsigned)B_SZ * NH * HD * S_LEN;
  if (i >= total) return;
  unsigned s = i % S_LEN; unsigned j = i / S_LEN;
  unsigned d = j % HD;    j /= HD;
  unsigned h = j % NH;    unsigned b = j / NH;
  vt[i] = qkvh[((size_t)(b * S_LEN + s)) * (3 * DMODEL) + 2 * DMODEL + h * HD + d];
}

// ---------------- bf16 WMMA GEMM ----------------
// C[M][N] = A[M][K] @ B[K][N], B supplied transposed: Bt[N][K].
// Block: 256 thr (8 waves) -> 256(M) x 64(N); wave -> 32(M) x 64(N), 8 accumulators.
// B tile (64x32) staged via async global->LDS copies, double-buffered; the next
// tile's async ops are issued right after the barrier so latency hides under WMMAs.
template <bool OUT_BF16>
__global__ __launch_bounds__(256) void gemm_bf16(const __bf16* __restrict__ A,
                                                 const __bf16* __restrict__ Bt,
                                                 void* __restrict__ C,
                                                 int M, int N, int K) {
  __shared__ __align__(16) __bf16 bsh[2][64 * 32];   // 8 KB double buffer
  const int tid  = threadIdx.x;
  const int lane = tid & 31;
  const int wave = __builtin_amdgcn_readfirstlane(tid >> 5);
  const int l16  = lane & 15;
  const int hi16 = lane >> 4;                 // 0 = lanes 0-15, 1 = lanes 16-31
  const int m0 = blockIdx.y * 256 + wave * 32;
  const int n0 = blockIdx.x * 64;

  v8f acc[8] = {};                            // [mi*4 + t]
  const __bf16* arow0 = A + (size_t)(m0 + l16) * K;
  const __bf16* arow1 = A + (size_t)(m0 + 16 + l16) * K;

  // cooperative B tile: 64 rows x 32 els; each thread one 16B chunk
  const int brow  = tid >> 2;                 // 0..63
  const int bcol8 = (tid & 3) * 8;            // 0,8,16,24
  const __bf16* bsrc = Bt + (size_t)(n0 + brow) * K + bcol8;
  const int bdst = brow * 32 + bcol8;

  async_ld_b128(bsh[0] + bdst, bsrc);         // prologue: tile k0 = 0

  for (int k0 = 0; k0 < K; k0 += 32) {
    wait_async0();                            // own tile-k0 async ops done
    __syncthreads();                          // all waves' writes visible
    if (k0 + 32 < K)                          // refill other buffer (safe post-barrier)
      async_ld_b128(bsh[((k0 >> 5) + 1) & 1] + bdst, bsrc + k0 + 32);
    const __bf16* buf = bsh[(k0 >> 5) & 1];

    const int koff = k0 + hi16 * 16;
    v16bf af0 = mk_frag(*(const v8bf*)(arow0 + koff), *(const v8bf*)(arow0 + koff + 8));
    v16bf af1 = mk_frag(*(const v8bf*)(arow1 + koff), *(const v8bf*)(arow1 + koff + 8));

#pragma unroll
    for (int t = 0; t < 4; ++t) {
      v16bf bf = *(const v16bf*)(buf + (t * 16 + l16) * 32 + hi16 * 16);
      acc[t]     = wmma_bf16(af0, bf, acc[t]);
      acc[4 + t] = wmma_bf16(af1, bf, acc[4 + t]);
    }
  }

#pragma unroll
  for (int mi = 0; mi < 2; ++mi) {
    const int rbase = m0 + mi * 16 + hi16 * 8;
#pragma unroll
    for (int t = 0; t < 4; ++t) {
      const int col = n0 + t * 16 + l16;
#pragma unroll
      for (int r = 0; r < 8; ++r) {
        float v = acc[mi * 4 + t][r];
        if (OUT_BF16)
          ((__bf16*)C)[(size_t)(rbase + r) * N + col] = f2bf(v);
        else
          ((float*)C)[(size_t)(rbase + r) * N + col] = v;
      }
    }
  }
}

// ---------------- flash attention ----------------
// grid(B*H, S/128), block 256 (8 waves). All 8 waves share one (b,h) and walk the
// same key blocks: K (32x64) and V^T (64x32) tiles staged via async global->LDS
// copies into double-buffered LDS (8x less global K/V traffic).
__global__ __launch_bounds__(256) void attn_kernel(const __bf16* __restrict__ qkvh,
                                                   const __bf16* __restrict__ vt,
                                                   __bf16* __restrict__ yh) {
  __shared__ __align__(16) __bf16 ksh[2][32 * 64];    // [buf][key][dim]  4 KB each
  __shared__ __align__(16) __bf16 vsh[2][64 * 32];    // [buf][dim][key]  4 KB each
  __shared__ __align__(16) __bf16 psh[8 * 16 * 32];   // wave-private P tiles
  const int tid  = threadIdx.x;
  const int lane = tid & 31;
  const int wave = __builtin_amdgcn_readfirstlane(tid >> 5);   // scalar
  const int l16  = lane & 15;
  const int hi16 = lane >> 4;
  const int bh = blockIdx.x;
  const int b = bh >> 4, h = bh & 15;
  const int qbase = (blockIdx.y * 8 + wave) * 16;              // scalar per wave
  const int kb_end     = (qbase + 15) >> 5;                    // this wave's range
  const int kb_blk_end = (blockIdx.y * 128 + 127) >> 5;        // block loop bound
  __bf16* pw = psh + wave * (16 * 32);

  // cooperative staging indices (one 16B chunk per thread per tile)
  const int krow_c = tid >> 3, kc8 = (tid & 7) * 8;   // K: 32 rows x 64 el
  const __bf16* ksrc = qkvh + (size_t)(b * S_LEN + krow_c) * (3 * DMODEL)
                            + DMODEL + h * HD + kc8;
  const int kdst = krow_c * 64 + kc8;
  const int vrow_c = tid >> 2, vc8 = (tid & 3) * 8;   // V^T: 64 rows x 32 el
  const __bf16* vsrc = vt + (size_t)bh * HD * S_LEN + (size_t)vrow_c * S_LEN + vc8;
  const int vdst = vrow_c * 32 + vc8;

  // Q fragments (head_dim 64 = 2 K-steps of 32)
  const __bf16* qrow = qkvh + (size_t)(b * S_LEN + qbase + l16) * (3 * DMODEL) + h * HD;
  v16bf qf[2];
#pragma unroll
  for (int ks = 0; ks < 2; ++ks) {
    const int koff = ks * 32 + hi16 * 16;
    qf[ks] = mk_frag(*(const v8bf*)(qrow + koff), *(const v8bf*)(qrow + koff + 8));
  }

  // all-ones B fragment: row-sums of P via one extra WMMA (l += P @ 1)
  v16bf ones;
  {
    const __bf16 one_bf = f2bf(1.0f);
#pragma unroll
    for (int i = 0; i < 16; ++i) ones[i] = one_bf;
  }

  float m[8];
  v8f o[4] = {};
  v8f lacc = {};
#pragma unroll
  for (int r = 0; r < 8; ++r) m[r] = -INFINITY;

  // prologue: async-stage tile 0
  async_ld_b128(ksh[0] + kdst, ksrc);
  async_ld_b128(vsh[0] + vdst, vsrc);

  for (int kb = 0; kb <= kb_blk_end; ++kb) {
    wait_async0();                             // own tile-kb ops complete
    __syncthreads();                           // everyone's tile-kb writes visible
    if (kb < kb_blk_end) {                     // refill other buffer (post-barrier safe)
      async_ld_b128(ksh[(kb + 1) & 1] + kdst,
                    ksrc + (size_t)(kb + 1) * 32 * (3 * DMODEL));
      async_ld_b128(vsh[(kb + 1) & 1] + vdst, vsrc + (kb + 1) * 32);
    }
    const __bf16* kbuf = ksh[kb & 1];
    const __bf16* vbuf = vsh[kb & 1];

    if (kb <= kb_end) {                        // scalar guard; EXEC stays full
      // ---- S = Q @ K^T : 4 score WMMAs from LDS fragments ----
      v16bf kf[2][2];
#pragma unroll
      for (int t = 0; t < 2; ++t)
#pragma unroll
        for (int ks = 0; ks < 2; ++ks)
          kf[t][ks] = *(const v16bf*)(kbuf + (t * 16 + l16) * 64 + ks * 32 + hi16 * 16);
      v8f s[2] = {};
#pragma unroll
      for (int t = 0; t < 2; ++t)
#pragma unroll
        for (int ks = 0; ks < 2; ++ks)
          s[t] = wmma_bf16(qf[ks], kf[t][ks], s[t]);

      // ---- V fragments from LDS; cheap, overlaps softmax ----
      v16bf vf[4];
#pragma unroll
      for (int t = 0; t < 4; ++t)
        vf[t] = *(const v16bf*)(vbuf + (t * 16 + l16) * 32 + hi16 * 16);

      // ---- scale (+ causal mask on diagonal blocks; scalar branch) ----
      if (kb * 32 + 31 <= qbase) {
#pragma unroll
        for (int t = 0; t < 2; ++t)
#pragma unroll
          for (int r = 0; r < 8; ++r) s[t][r] *= 0.125f;
      } else {
#pragma unroll
        for (int t = 0; t < 2; ++t) {
          const int j = kb * 32 + t * 16 + l16;
#pragma unroll
          for (int r = 0; r < 8; ++r) {
            const int i = qbase + hi16 * 8 + r;
            float v = s[t][r] * 0.125f;
            s[t][r] = (j <= i) ? v : -INFINITY;
          }
        }
      }

      // ---- online softmax: row max via 4 shuffles; stage P into LDS ----
#pragma unroll
      for (int r = 0; r < 8; ++r) {
        float v = fmaxf(s[0][r], s[1][r]);
        v = fmaxf(v, __shfl_xor(v, 1, 32));
        v = fmaxf(v, __shfl_xor(v, 2, 32));
        v = fmaxf(v, __shfl_xor(v, 4, 32));
        v = fmaxf(v, __shfl_xor(v, 8, 32));    // max across 16 lanes (same half)
        float mnew = fmaxf(m[r], v);
        float corr = __expf(m[r] - mnew);
        float p0 = __expf(s[0][r] - mnew);
        float p1 = __expf(s[1][r] - mnew);
        m[r] = mnew;
        lacc[r] *= corr;
#pragma unroll
        for (int t = 0; t < 4; ++t) o[t][r] *= corr;
        const int prow = hi16 * 8 + r;
        pw[prow * 32 + l16]      = f2bf(p0);
        pw[prow * 32 + 16 + l16] = f2bf(p1);
      }
      asm volatile("s_wait_dscnt 0" ::: "memory");  // P visible before frag read

      // ---- read P back as 16x32 A fragment ----
      const int pko = hi16 * 16;
      v16bf pf = mk_frag(*(const v8bf*)(pw + l16 * 32 + pko),
                         *(const v8bf*)(pw + l16 * 32 + pko + 8));

      // ---- O += P @ V ; l += P @ 1 (row sums already in C/D layout) ----
#pragma unroll
      for (int t = 0; t < 4; ++t) o[t] = wmma_bf16(pf, vf[t], o[t]);
      lacc = wmma_bf16(pf, ones, lacc);
    }
  }

  // ---- normalize and store y (bf16, [B*S][D], heads interleaved) ----
#pragma unroll
  for (int t = 0; t < 4; ++t) {
    const int d = t * 16 + l16;
#pragma unroll
    for (int r = 0; r < 8; ++r) {
      const int srow = qbase + hi16 * 8 + r;
      float v = o[t][r] / lacc[r];
      yh[(size_t)(b * S_LEN + srow) * DMODEL + h * HD + d] = f2bf(v);
    }
  }
}

// ---------------- host launch ----------------
extern "C" void kernel_launch(void* const* d_in, const int* in_sizes, int n_in,
                              void* d_out, int out_size, void* d_ws, size_t ws_size,
                              hipStream_t stream) {
  const float* x      = (const float*)d_in[0];   // [2,2048,1024]
  const float* w_qkv  = (const float*)d_in[1];   // [1024,3072]
  const float* w_proj = (const float*)d_in[2];   // [1024,1024]
  float* out = (float*)d_out;

  char* ws = (char*)d_ws;
  __bf16* xh     = (__bf16*)(ws + 0);          //  8 MB : x bf16 [4096][1024]
  __bf16* wqkvt  = (__bf16*)(ws + 8388608);    //  6 MB : w_qkv^T bf16 [3072][1024]
  __bf16* wprojt = (__bf16*)(ws + 14680064);   //  2 MB : w_proj^T bf16 [1024][1024]
  __bf16* qkvh   = (__bf16*)(ws + 16777216);   // 24 MB : qkv bf16 [4096][3072]
  __bf16* vt     = (__bf16*)(ws + 41943040);   //  8 MB : V^T bf16 [32][64][2048]
  __bf16* yh     = (__bf16*)(ws + 50331648);   //  8 MB : attn out bf16 [4096][1024]

  const unsigned NX = 4096u * 1024u;
  cvt_f32_bf16<<<(NX + 255) / 256, 256, 0, stream>>>(x, xh, NX);
  transpose_f32_to_bf16<<<(1024u * 3072u + 255) / 256, 256, 0, stream>>>(w_qkv, wqkvt, 1024, 3072);
  transpose_f32_to_bf16<<<(1024u * 1024u + 255) / 256, 256, 0, stream>>>(w_proj, wprojt, 1024, 1024);

  // qkv = x @ w_qkv  (M=4096, N=3072, K=1024)
  gemm_bf16<true><<<dim3(3072 / 64, 4096 / 256), 256, 0, stream>>>(
      xh, wqkvt, (void*)qkvh, 4096, 3072, 1024);

  build_vt<<<(2u * 16u * 64u * 2048u + 255) / 256, 256, 0, stream>>>(qkvh, vt);

  // flash attention: grid(B*H, S/128)
  attn_kernel<<<dim3(B_SZ * NH, S_LEN / 128), 256, 0, stream>>>(qkvh, vt, yh);

  // out = y @ w_proj  (M=4096, N=1024, K=1024)
  gemm_bf16<false><<<dim3(1024 / 64, 4096 / 256), 256, 0, stream>>>(
      yh, wprojt, (void*)out, 4096, 1024, 1024);
}
